// GLAMemoryReader_7456063226408
// MI455X (gfx1250) — compile-verified, matrix operand order checked by reading, not run on hardware
//
#include <hip/hip_runtime.h>
#include <hip/hip_bf16.h>
#include <math.h>

// ---------------------------------------------------------------------------
// GLA memory reader for MI455X (gfx1250, wave32, WMMA).
// Pipeline: cvt weights->bf16 | rmsnorm(x)->bf16 | WMMA GEMM q | fused
// (L2norm -> einsum WMMA -> v-rmsnorm) | WMMA GEMM o-proj | WMMA GEMM gate
// with fused sigmoid*o epilogue.
// GEMM: 256x128x32 block tile, 64x64 wave tile (16 WMMA / 16 ds_load_b128),
// double-buffered LDS staging via GLOBAL_LOAD_ASYNC_TO_LDS_B128 (ASYNCcnt).
// ---------------------------------------------------------------------------

typedef __attribute__((ext_vector_type(16))) __bf16 v16bf;
typedef __attribute__((ext_vector_type(8)))  __bf16 v8bf;
typedef __attribute__((ext_vector_type(4)))  __bf16 v4bf;
typedef __attribute__((ext_vector_type(8)))  float  v8f;

#define MTOK   32768   // B*T
#define DHID   2048
#define KEYD   1024
#define NHEAD  16
#define HEADK  64
#define HEADV  128
#define EPSF   1e-5f

static __device__ __forceinline__ __bf16 to_bf16(float x) { return (__bf16)x; }

// CDNA5 async copy: global -> LDS, 16B per lane, tracked by ASYNCcnt.
static __device__ __forceinline__ void async_b128(unsigned lds_byte, const void* g) {
  asm volatile("global_load_async_to_lds_b128 %0, %1, off"
               :: "v"(lds_byte), "v"(g) : "memory");
}
static __device__ __forceinline__ void wait_async0() {
  asm volatile("s_wait_asynccnt 0x0" ::: "memory");
}
// low 32 bits of a generic pointer into the shared aperture == LDS byte address
static __device__ __forceinline__ unsigned lds_addr(const void* p) {
  return (unsigned)(unsigned long long)p;
}

// ---------------------------------------------------------------------------
// fp32 -> bf16 bulk convert
// ---------------------------------------------------------------------------
__global__ __launch_bounds__(256) void cvt_f32_bf16(const float* __restrict__ in,
                                                    unsigned short* __restrict__ out,
                                                    long n4) {
  long i = (long)blockIdx.x * 256 + threadIdx.x;
  if (i >= n4) return;
  float4 v = ((const float4*)in)[i];
  v4bf o;
  o[0] = to_bf16(v.x); o[1] = to_bf16(v.y);
  o[2] = to_bf16(v.z); o[3] = to_bf16(v.w);
  ((v4bf*)out)[i] = o;
}

// ---------------------------------------------------------------------------
// RMSNorm over D=2048, write bf16. One block (256 thr) per row.
// ---------------------------------------------------------------------------
__global__ __launch_bounds__(256) void rmsnorm_x_kernel(const float* __restrict__ x,
                                                        const float* __restrict__ w,
                                                        unsigned short* __restrict__ out) {
  __shared__ float red[8];
  const size_t row = blockIdx.x;
  const int tid = threadIdx.x;
  const float4* xr = (const float4*)(x + row * DHID);
  const float4* wr = (const float4*)w;
  float4 v0 = xr[tid];
  float4 v1 = xr[tid + 256];
  float s = v0.x*v0.x + v0.y*v0.y + v0.z*v0.z + v0.w*v0.w
          + v1.x*v1.x + v1.y*v1.y + v1.z*v1.z + v1.w*v1.w;
  #pragma unroll
  for (int o = 16; o > 0; o >>= 1) s += __shfl_xor(s, o, 32);
  if ((tid & 31) == 0) red[tid >> 5] = s;
  __syncthreads();
  float tot = 0.f;
  #pragma unroll
  for (int i = 0; i < 8; ++i) tot += red[i];
  const float sc = rsqrtf(tot * (1.0f / DHID) + EPSF);
  float4 w0 = wr[tid], w1 = wr[tid + 256];
  v4bf o0, o1;
  o0[0] = to_bf16(v0.x * sc * w0.x); o0[1] = to_bf16(v0.y * sc * w0.y);
  o0[2] = to_bf16(v0.z * sc * w0.z); o0[3] = to_bf16(v0.w * sc * w0.w);
  o1[0] = to_bf16(v1.x * sc * w1.x); o1[1] = to_bf16(v1.y * sc * w1.y);
  o1[2] = to_bf16(v1.z * sc * w1.z); o1[3] = to_bf16(v1.w * sc * w1.w);
  v4bf* op = (v4bf*)(out + row * DHID);
  op[tid] = o0;
  op[tid + 256] = o1;
}

// ---------------------------------------------------------------------------
// Tiled bf16 WMMA GEMM: out[M,N] = A[M,K] @ W[N,K]^T  (fp32 accumulate)
// BM=256, BN=128, BK=32; 256 threads = 8 waves; wave tile 64x64 (4x4 frags).
// mode 0: out[idx] = acc
// mode 1: out[idx] = din[idx] * sigmoid(acc)     (gate fusion)
// ---------------------------------------------------------------------------
#define GBM 256
#define GBN 128
#define GBK 32
#define GSA 40                     // padded LDS stride in halves (80B rows)
#define ABUF  (GBM * GSA)          // halves per A buffer
#define BBUF  (GBN * GSA)          // halves per B buffer
#define ABUFB (ABUF * 2)           // bytes
#define BBUFB (BBUF * 2)

static __device__ __forceinline__ void stage_tile_async(const __bf16* gaK, const __bf16* gbK,
                                                        unsigned aO, unsigned bO, int K) {
  async_b128(aO,                  gaK);
  async_b128(aO +  64u * GSA * 2, gaK + (size_t)64  * K);
  async_b128(aO + 128u * GSA * 2, gaK + (size_t)128 * K);
  async_b128(aO + 192u * GSA * 2, gaK + (size_t)192 * K);
  async_b128(bO,                  gbK);
  async_b128(bO +  64u * GSA * 2, gbK + (size_t)64  * K);
}

__global__ __launch_bounds__(256) void gemm_bf16_wmma(const unsigned short* __restrict__ Au,
                                                      const unsigned short* __restrict__ Wu,
                                                      float* __restrict__ out,
                                                      const float* __restrict__ din,
                                                      int M, int N, int K, int mode) {
  __shared__ __bf16 As[2 * ABUF];
  __shared__ __bf16 Bs[2 * BBUF];
  const __bf16* A = (const __bf16*)Au;
  const __bf16* W = (const __bf16*)Wu;

  const int tid  = threadIdx.x;
  const int lane = tid & 31;
  const int wv   = tid >> 5;         // 0..7
  const int wm   = (wv & 3) * 64;    // 4 waves along M
  const int wn   = (wv >> 2) * 64;   // 2 waves along N
  const int lm   = lane & 15;
  const int hi   = lane >> 4;        // 0 or 1 (lane half)
  const int mBase = blockIdx.y * GBM;
  const int nBase = blockIdx.x * GBN;

  // per-thread staging slot: chunk = tid -> row tid>>2, kchunk (tid&3)*8 halves
  const int srow = tid >> 2;
  const int skc  = (tid & 3) * 8;
  const __bf16* ga = A + (size_t)(mBase + srow) * K + skc;
  const __bf16* gb = W + (size_t)(nBase + srow) * K + skc;
  const unsigned ldsSlot = (unsigned)((srow * GSA + skc) * 2);
  const unsigned aBase = lds_addr(As) + ldsSlot;
  const unsigned bBase = lds_addr(Bs) + ldsSlot;

  v8f acc[4][4];
  #pragma unroll
  for (int mi = 0; mi < 4; ++mi)
    #pragma unroll
    for (int ni = 0; ni < 4; ++ni)
      acc[mi][ni] = (v8f)(0.0f);

  const int numK = K / GBK;
  // prologue: prefetch tile 0 into buffer 0
  stage_tile_async(ga, gb, aBase, bBase, K);

  int p = 0;
  for (int kt = 0; kt < numK; ++kt) {
    wait_async0();        // my async writes into buffer p have landed
    __syncthreads();      // everyone's writes visible; prior reads of p^1 done

    // branchless prefetch of the next tile into the other buffer (last
    // iteration harmlessly re-fetches the current tile; S_ENDPGM's implicit
    // wait-idle covers the dangling ASYNCcnt).
    const int knext = (kt + 1 < numK) ? (kt + 1) : kt;
    stage_tile_async(ga + (size_t)knext * GBK, gb + (size_t)knext * GBK,
                     aBase + (unsigned)((p ^ 1) * ABUFB),
                     bBase + (unsigned)((p ^ 1) * BBUFB), K);

    const __bf16* Ab = As + p * ABUF;
    const __bf16* Bb = Bs + p * BBUF;

    // 16-bit A layout: lane(lo) k {0..7,16..23}, lane(hi) k {8..15,24..31}
    v16bf af[4], bfv[4];
    #pragma unroll
    for (int mi = 0; mi < 4; ++mi) {
      int r = wm + mi * 16 + lm;
      v8bf lo = *(const v8bf*)(Ab + r * GSA + hi * 8);
      v8bf hh = *(const v8bf*)(Ab + r * GSA + hi * 8 + 16);
      af[mi] = __builtin_shufflevector(lo, hh, 0,1,2,3,4,5,6,7,8,9,10,11,12,13,14,15);
    }
    #pragma unroll
    for (int ni = 0; ni < 4; ++ni) {
      int r = wn + ni * 16 + lm;
      v8bf lo = *(const v8bf*)(Bb + r * GSA + hi * 8);
      v8bf hh = *(const v8bf*)(Bb + r * GSA + hi * 8 + 16);
      bfv[ni] = __builtin_shufflevector(lo, hh, 0,1,2,3,4,5,6,7,8,9,10,11,12,13,14,15);
    }

    #pragma unroll
    for (int mi = 0; mi < 4; ++mi)
      #pragma unroll
      for (int ni = 0; ni < 4; ++ni)
        acc[mi][ni] = __builtin_amdgcn_wmma_f32_16x16x32_bf16(
            false, af[mi], false, bfv[ni], (short)0, acc[mi][ni], false, false);

    p ^= 1;
  }

  // --- epilogue: C layout = VGPR r -> M = r + hi*8, lanes%16 -> N ---
  const float* dbase = (mode == 1) ? din : out;
  #pragma unroll
  for (int mi = 0; mi < 4; ++mi) {
    #pragma unroll
    for (int ni = 0; ni < 4; ++ni) {
      const int col  = nBase + wn + ni * 16 + lm;
      const int row0 = mBase + wm + mi * 16 + hi * 8;
      float* po       = out   + (size_t)row0 * N + col;
      const float* pi = dbase + (size_t)row0 * N + col;
      if (mode == 0) {
        #pragma unroll
        for (int r = 0; r < 8; ++r) po[(size_t)r * N] = acc[mi][ni][r];
      } else {
        #pragma unroll
        for (int r = 0; r < 8; ++r) {
          const float g = __builtin_amdgcn_rcpf(1.0f + __expf(-acc[mi][ni][r]));
          po[(size_t)r * N] = pi[(size_t)r * N] * g;
        }
      }
    }
  }
}

// ---------------------------------------------------------------------------
// Fused: L2-normalize q per head (K=64) -> o = q @ S[b,h] (64x128 WMMA)
//        -> RMSNorm over V=128 -> bf16 o-vector.
// Block: 256 thr = 8 waves, each wave does 16 tokens; grid (M/128, H).
// ---------------------------------------------------------------------------
#define SSK 72      // padded LDS stride (halves) for transposed S tile

__global__ __launch_bounds__(256) void q_state_kernel(const float* __restrict__ q,
                                                      const unsigned short* __restrict__ Su,
                                                      const float* __restrict__ vnw,
                                                      unsigned short* __restrict__ obf) {
  __shared__ __bf16 Sl[HEADV * SSK];          // S^T : [v][k]
  const __bf16* S = (const __bf16*)Su;
  const int h    = blockIdx.y;
  const int tok0 = blockIdx.x * 128;
  const int b    = tok0 >> 12;                // T = 4096
  const __bf16* Sp = S + ((size_t)(b * NHEAD + h)) * HEADK * HEADV;

  // stage S (64x128) transposed into LDS
  #pragma unroll
  for (int i = 0; i < 4; ++i) {
    int c  = threadIdx.x + i * 256;           // 0..1023 chunks of 8 along v
    int k  = c >> 4;
    int vc = (c & 15) * 8;
    v8bf d = *(const v8bf*)(Sp + k * HEADV + vc);
    #pragma unroll
    for (int e = 0; e < 8; ++e) Sl[(vc + e) * SSK + k] = d[e];
  }
  __syncthreads();

  const int lane = threadIdx.x & 31;
  const int wv   = threadIdx.x >> 5;
  const int lm   = lane & 15;
  const int hi   = lane >> 4;
  const int tw0  = tok0 + wv * 16;
  const int tok  = tw0 + lm;                  // A-matrix row for this lane
  const float* qp = q + (size_t)tok * KEYD + h * HEADK;

  // load this lane's 32 q values (A-frag k pattern), L2-normalize the row
  float f[2][16];
  float ss = 0.f;
  #pragma unroll
  for (int s = 0; s < 2; ++s) {
    const int ks = s * 32 + hi * 8;
    #pragma unroll
    for (int e = 0; e < 8; ++e) { f[s][e] = qp[ks + e]; f[s][8 + e] = qp[ks + 16 + e]; }
    #pragma unroll
    for (int e = 0; e < 16; ++e) ss += f[s][e] * f[s][e];
  }
  ss += __shfl_xor(ss, 16, 32);               // lanes L and L+16 hold the row
  const float inv = 1.0f / fmaxf(sqrtf(ss), 1e-12f);

  v16bf a[2];
  #pragma unroll
  for (int s = 0; s < 2; ++s)
    #pragma unroll
    for (int e = 0; e < 16; ++e) a[s][e] = to_bf16(f[s][e] * inv);

  // 8 N-fragments of 16 -> full V=128; K=64 via two k-steps
  v8f acc[8];
  #pragma unroll
  for (int j = 0; j < 8; ++j) acc[j] = (v8f)(0.0f);
  #pragma unroll
  for (int j = 0; j < 8; ++j) {
    const int n = j * 16 + lm;
    #pragma unroll
    for (int s = 0; s < 2; ++s) {
      const int kb = s * 32 + hi * 8;
      v8bf lo = *(const v8bf*)(Sl + n * SSK + kb);
      v8bf hh = *(const v8bf*)(Sl + n * SSK + kb + 16);
      v16bf bfrag = __builtin_shufflevector(lo, hh, 0,1,2,3,4,5,6,7,8,9,10,11,12,13,14,15);
      acc[j] = __builtin_amdgcn_wmma_f32_16x16x32_bf16(
          false, a[s], false, bfrag, (short)0, acc[j], false, false);
    }
  }

  // per-row RMSNorm over V=128: row M=r+hi*8 lives entirely in this 16-lane half
  float vw[8];
  #pragma unroll
  for (int j = 0; j < 8; ++j) vw[j] = vnw[j * 16 + lm];
  #pragma unroll
  for (int r = 0; r < 8; ++r) {
    float s2 = 0.f;
    #pragma unroll
    for (int j = 0; j < 8; ++j) s2 += acc[j][r] * acc[j][r];
    s2 += __shfl_xor(s2, 1, 32);
    s2 += __shfl_xor(s2, 2, 32);
    s2 += __shfl_xor(s2, 4, 32);
    s2 += __shfl_xor(s2, 8, 32);
    const float sc = rsqrtf(s2 * (1.0f / HEADV) + EPSF);
    const int trow = tw0 + hi * 8 + r;
    __bf16* op = (__bf16*)obf + (size_t)trow * DHID + h * HEADV;
    #pragma unroll
    for (int j = 0; j < 8; ++j) op[j * 16 + lm] = to_bf16(acc[j][r] * sc * vw[j]);
  }
}

// ---------------------------------------------------------------------------
// Host-side orchestration
// ---------------------------------------------------------------------------
extern "C" void kernel_launch(void* const* d_in, const int* in_sizes, int n_in,
                              void* d_out, int out_size, void* d_ws, size_t ws_size,
                              hipStream_t stream) {
  const float* hs  = (const float*)d_in[0];   // (B,T,2048)
  const float* rs  = (const float*)d_in[1];   // (B,16,64,128)
  const float* nw  = (const float*)d_in[2];   // (2048)
  const float* qw  = (const float*)d_in[3];   // (1024,2048)
  const float* vnw = (const float*)d_in[4];   // (128)
  const float* ow  = (const float*)d_in[5];   // (2048,2048)
  const float* gw  = (const float*)d_in[6];   // (2048,2048)
  float* out = (float*)d_out;

  char* ws = (char*)d_ws;
  size_t off = 0;
  auto take = [&](size_t bytes) -> void* {
    void* p = ws + off;
    off = (off + bytes + 255) & ~(size_t)255;
    return p;
  };
  unsigned short* xbf  = (unsigned short*)take((size_t)MTOK * DHID * 2);   // 128 MB
  unsigned short* qwbf = (unsigned short*)take((size_t)KEYD * DHID * 2);   //   4 MB
  unsigned short* owbf = (unsigned short*)take((size_t)DHID * DHID * 2);   //   8 MB
  unsigned short* gwbf = (unsigned short*)take((size_t)DHID * DHID * 2);   //   8 MB
  unsigned short* sbf  = (unsigned short*)take((size_t)8 * NHEAD * HEADK * HEADV * 2); // 2 MB
  float*          qf   = (float*)take((size_t)MTOK * KEYD * 4);            // 128 MB
  unsigned short* obfp = (unsigned short*)take((size_t)MTOK * DHID * 2);   // 128 MB

  // 1) weights + state -> bf16
  cvt_f32_bf16<<<(KEYD * DHID / 4 + 255) / 256, 256, 0, stream>>>(qw, qwbf, (long)KEYD * DHID / 4);
  cvt_f32_bf16<<<(DHID * DHID / 4 + 255) / 256, 256, 0, stream>>>(ow, owbf, (long)DHID * DHID / 4);
  cvt_f32_bf16<<<(DHID * DHID / 4 + 255) / 256, 256, 0, stream>>>(gw, gwbf, (long)DHID * DHID / 4);
  cvt_f32_bf16<<<(8 * NHEAD * HEADK * HEADV / 4 + 255) / 256, 256, 0, stream>>>(
      rs, sbf, (long)8 * NHEAD * HEADK * HEADV / 4);

  // 2) x = rmsnorm(hidden) -> bf16
  rmsnorm_x_kernel<<<MTOK, 256, 0, stream>>>(hs, nw, xbf);

  // 3) q = x @ q_w^T  (fp32 out, normalized in the next kernel)
  gemm_bf16_wmma<<<dim3(KEYD / GBN, MTOK / GBM), 256, 0, stream>>>(
      xbf, qwbf, qf, nullptr, MTOK, KEYD, DHID, 0);

  // 4) L2norm(q) -> einsum with recurrent state -> rmsnorm_V -> o (bf16)
  q_state_kernel<<<dim3(MTOK / 128, NHEAD), 256, 0, stream>>>(qf, sbf, vnw, obfp);

  // 5) o_proj = o @ o_w^T -> d_out (fp32)
  gemm_bf16_wmma<<<dim3(DHID / GBN, MTOK / GBM), 256, 0, stream>>>(
      obfp, owbf, out, nullptr, MTOK, DHID, DHID, 0);

  // 6) out = sigmoid(x @ gate_w^T) * d_out   (fused epilogue)
  gemm_bf16_wmma<<<dim3(DHID / GBN, MTOK / GBM), 256, 0, stream>>>(
      xbf, gwbf, out, out, MTOK, DHID, DHID, 1);
}
